// Attention_24068996727489
// MI455X (gfx1250) — matrix operand order repaired and checked by reading, not compiled, not run on hardware
//
#include <hip/hip_runtime.h>
#include <hip/hip_bf16.h>

// ---------------------------------------------------------------------------
// Types
// ---------------------------------------------------------------------------
typedef __bf16 bf16_t;
typedef __bf16 bf16x16 __attribute__((ext_vector_type(16)));
typedef float  f32x8   __attribute__((ext_vector_type(8)));
typedef float  f32x4   __attribute__((ext_vector_type(4)));
typedef unsigned int u32x4 __attribute__((ext_vector_type(4)));
typedef int i32x4 __attribute__((ext_vector_type(4)));

typedef __attribute__((address_space(1))) i32x4 g_i32x4;
typedef __attribute__((address_space(3))) i32x4 l_i32x4;

#define NUM_HEADS 32
#define SEQ 2048
#define DMODEL 4096
#define HEADDIM 128

// ---------------------------------------------------------------------------
// WMMA fragment helpers (wave32, v_wmma_f32_16x16x32_bf16)
// A 16x32: lanes hold M=lane&15; lane<16 -> K{0..7,16..23}, lane>=16 -> K{8..15,24..31}
// B 32x16: lane holds col N=lane&15; lane<16 -> K{0..7,16..23}, lane>=16 -> K{8..15,24..31}
//          (B staged in LDS as [n][k] so each lane reads contiguous bytes)
// C/D 16x16 f32: VGPR r = row r (lanes 0-15) / row r+8 (lanes 16-31), col = lane&15
// ---------------------------------------------------------------------------
__device__ __forceinline__ bf16x16 load_frag_a(const bf16_t* base, int ld,
                                               int m0, int k0, int lane) {
    int half = lane >> 4;
    int mr   = lane & 15;
    const bf16_t* p = base + (m0 + mr) * ld + k0 + half * 8;
    union { u32x4 q[2]; bf16x16 v; } f;
    f.q[0] = *(const u32x4*)(p);
    f.q[1] = *(const u32x4*)(p + 16);
    return f.v;
}

__device__ __forceinline__ bf16x16 load_frag_b(const bf16_t* base, int ld,
                                               int n0, int k0, int lane) {
    int half = lane >> 4;
    int nr   = lane & 15;
    const bf16_t* p = base + (n0 + nr) * ld + k0 + half * 8;
    union { u32x4 q[2]; bf16x16 v; } f;
    f.q[0] = *(const u32x4*)(p);
    f.q[1] = *(const u32x4*)(p + 16);
    return f.v;
}

__device__ __forceinline__ f32x8 wmma_bf16(bf16x16 a, bf16x16 b, f32x8 c) {
    return __builtin_amdgcn_wmma_f32_16x16x32_bf16(false, a, false, b,
                                                   (short)0, c, false, false);
}

// Async global -> LDS copy of 16 bytes (gfx1250 GLOBAL_LOAD_ASYNC_TO_LDS_B128)
__device__ __forceinline__ void async_b128(const bf16_t* g, bf16_t* l) {
    __builtin_amdgcn_global_load_async_to_lds_b128((g_i32x4*)g, (l_i32x4*)l, 0, 0);
}

// ---------------------------------------------------------------------------
// fp32 -> bf16 elementwise convert (for x)
// ---------------------------------------------------------------------------
__global__ __launch_bounds__(256)
void cvt_f32_bf16(const float* __restrict__ X, bf16_t* __restrict__ Y) {
    const int i = (blockIdx.x * 256 + threadIdx.x) * 8;
    f32x4 a = *(const f32x4*)(X + i);
    f32x4 b = *(const f32x4*)(X + i + 4);
    union { u32x4 u; bf16_t h[8]; } o;
    #pragma unroll
    for (int j = 0; j < 4; ++j) { o.h[j] = (bf16_t)a[j]; o.h[4 + j] = (bf16_t)b[j]; }
    *(u32x4*)(Y + i) = o.u;
}

// ---------------------------------------------------------------------------
// W[K][N] fp32 -> Wt[N][K] bf16  (LDS-tiled transpose + convert, 64x64 tiles)
// ---------------------------------------------------------------------------
__global__ __launch_bounds__(256)
void transpose_cvt(const float* __restrict__ W, bf16_t* __restrict__ Wt,
                   int K, int N) {
    __shared__ __align__(16) bf16_t t[64 * 72];       // [n][k], ld=72
    const int k0 = blockIdx.x * 64;
    const int n0 = blockIdx.y * 64;
    #pragma unroll
    for (int it = 0; it < 4; ++it) {                   // 64 rows * 16 f32x4
        const int i  = threadIdx.x + it * 256;
        const int k  = i >> 4;
        const int c4 = i & 15;
        f32x4 f = *(const f32x4*)(W + (size_t)(k0 + k) * N + n0 + c4 * 4);
        const int n = c4 * 4;
        t[(n + 0) * 72 + k] = (bf16_t)f[0];
        t[(n + 1) * 72 + k] = (bf16_t)f[1];
        t[(n + 2) * 72 + k] = (bf16_t)f[2];
        t[(n + 3) * 72 + k] = (bf16_t)f[3];
    }
    __syncthreads();
    #pragma unroll
    for (int it = 0; it < 2; ++it) {                   // 64 rows * 8 b128
        const int i  = threadIdx.x + it * 256;
        const int n  = i >> 3;
        const int k8 = (i & 7) * 8;
        u32x4 v = *(const u32x4*)(t + n * 72 + k8);
        *(u32x4*)(Wt + (size_t)(n0 + n) * K + k0 + k8) = v;
    }
}

// ---------------------------------------------------------------------------
// GEMM: C[M,N] = A[M,K] * Bt[N,K]^T, bf16 inputs, fp32 accumulate.
// Tile 128x128, K-step 32, 256 threads (8 waves), wave = 32x64 sub-tile.
// Double-buffered LDS filled by GLOBAL_LOAD_ASYNC_TO_LDS_B128 (ASYNCcnt).
// Per wave, each K-batch = 4 async ops (2 for A + 2 for B).
// MODE 0: fp32 out; 1: bf16 out; 2: bf16 out + fused RoPE.
// ---------------------------------------------------------------------------
#define GBM 128
#define GBN 128
#define GBK 32
#define LDK 40    // padded K stride: 40*2B = 80B (16B aligned)

template <int MODE>
__global__ __launch_bounds__(256)
void gemm_bf16(const bf16_t* __restrict__ A, const bf16_t* __restrict__ Bt,
               void* __restrict__ Cout, const float* __restrict__ cosb,
               const float* __restrict__ sinb, int M, int N, int K) {
    __shared__ __align__(16) bf16_t ldsA[2][GBM * LDK];
    __shared__ __align__(16) bf16_t ldsB[2][GBN * LDK];

    const int tid  = threadIdx.x;
    const int lane = tid & 31;
    const int wave = tid >> 5;
    const int m_blk = blockIdx.y * GBM;
    const int n_blk = blockIdx.x * GBN;
    const int wm = (wave & 3) * 32;
    const int wn = (wave >> 2) * 64;

    // chunk map: tile 128 rows x 4 b128-chunks (32 bf16), 2 chunks/thread
    int arow[2], acol[2];
    #pragma unroll
    for (int it = 0; it < 2; ++it) {
        const int c = tid + it * 256;
        arow[it] = c >> 2;
        acol[it] = (c & 3) * 8;
    }
    const bf16_t* Abase = A  + (size_t)m_blk * K;
    const bf16_t* Bbase = Bt + (size_t)n_blk * K;

    // issue one K-batch of async copies into buffer `buf`
    auto issue = [&](int buf, int kt) {
        #pragma unroll
        for (int it = 0; it < 2; ++it) {
            async_b128(Abase + (size_t)arow[it] * K + kt + acol[it],
                       &ldsA[buf][arow[it] * LDK + acol[it]]);
            async_b128(Bbase + (size_t)arow[it] * K + kt + acol[it],
                       &ldsB[buf][arow[it] * LDK + acol[it]]);
        }
    };

    issue(0, 0);
    issue(1, GBK);

    f32x8 acc[2][4] = {};

    int i = 0;
    for (int kt = 0; kt < K; kt += GBK, ++i) {
        const int cur = i & 1;
        // wait until only the next batch (4 async ops) is still outstanding
        if (kt + GBK < K) __builtin_amdgcn_s_wait_asynccnt(4);
        else              __builtin_amdgcn_s_wait_asynccnt(0);
        __syncthreads();

        #pragma unroll
        for (int mi = 0; mi < 2; ++mi) {
            bf16x16 a = load_frag_a(&ldsA[cur][0], LDK, wm + mi * 16, 0, lane);
            #pragma unroll
            for (int ni = 0; ni < 4; ++ni) {
                bf16x16 b = load_frag_b(&ldsB[cur][0], LDK, wn + ni * 16, 0, lane);
                acc[mi][ni] = wmma_bf16(a, b, acc[mi][ni]);
            }
        }
        __syncthreads();
        if (kt + 2 * GBK < K) issue(cur, kt + 2 * GBK);
    }

    // ---- epilogue ----
    const int half = lane >> 4;
    const int nr   = lane & 15;
    #pragma unroll
    for (int mi = 0; mi < 2; ++mi) {
        #pragma unroll
        for (int ni = 0; ni < 4; ++ni) {
            const int col  = n_blk + wn + ni * 16 + nr;
            const int row0 = m_blk + wm + mi * 16 + half * 8;
            #pragma unroll
            for (int r = 0; r < 8; ++r) {
                float v = acc[mi][ni][r];
                if constexpr (MODE == 2) {
                    // RoPE: partner column lives in the adjacent lane
                    const float part = __shfl_xor(v, 1, 32);
                    const int   p    = (col & (HEADDIM - 1)) >> 1;
                    const float c    = cosb[(row0 + r) * (HEADDIM / 2) + p];
                    const float sn   = sinb[(row0 + r) * (HEADDIM / 2) + p];
                    v = (col & 1) ? (part * sn + v * c) : (v * c - part * sn);
                }
                if constexpr (MODE == 0) {
                    ((float*)Cout)[(size_t)(row0 + r) * N + col] = v;
                } else {
                    ((bf16_t*)Cout)[(size_t)(row0 + r) * N + col] = (bf16_t)v;
                }
            }
        }
    }
}

// ---------------------------------------------------------------------------
// Flash attention (non-causal). Block = (1 head, 64 query rows), 128 thr = 4 waves.
// bf16 Q/K/V in [S][D]; bf16 output. K/V blocks use register-prefetch pipeline.
// ---------------------------------------------------------------------------
#define QT 64
#define KT 64
#define LQ (HEADDIM + 8)  // 136
#define LK (HEADDIM + 8)
#define LV (KT + 8)       // 72, ldsV transposed [d][key]
#define LP (KT + 8)

__global__ __launch_bounds__(128)
void attn_kernel(const bf16_t* __restrict__ qb, const bf16_t* __restrict__ kbp,
                 const bf16_t* __restrict__ vb, bf16_t* __restrict__ ab) {
    __shared__ __align__(16) bf16_t ldsQ[QT * LQ];
    __shared__ __align__(16) bf16_t ldsK[KT * LK];
    __shared__ __align__(16) bf16_t ldsV[HEADDIM * LV];
    __shared__ __align__(16) bf16_t ldsP[4 * 16 * LP];

    const int tid  = threadIdx.x;
    const int lane = tid & 31;
    const int wave = tid >> 5;
    const int h    = blockIdx.y;
    const int q0   = blockIdx.x * QT;
    const size_t headOff = (size_t)h * HEADDIM;

    // ---- stage Q tile 64x128 (b128 copies) ----
    #pragma unroll
    for (int it = 0; it < 8; ++it) {
        const int i  = tid + it * 128;
        const int r  = i >> 4;
        const int c8 = (i & 15) * 8;
        *(u32x4*)(ldsQ + r * LQ + c8) =
            *(const u32x4*)(qb + (size_t)(q0 + r) * DMODEL + headOff + c8);
    }

    // K/V prefetch registers (8 b128 chunks each)
    u32x4 rk[8], rv[8];
    auto preload = [&](int kb) {
        #pragma unroll
        for (int it = 0; it < 8; ++it) {
            const int i  = tid + it * 128;
            const int r  = i >> 4;
            const int c8 = (i & 15) * 8;
            rk[it] = *(const u32x4*)(kbp + (size_t)(kb + r) * DMODEL + headOff + c8);
            rv[it] = *(const u32x4*)(vb  + (size_t)(kb + r) * DMODEL + headOff + c8);
        }
    };
    preload(0);

    f32x8 oacc[8] = {};
    float mrow[8], lrow[8];
    #pragma unroll
    for (int r = 0; r < 8; ++r) { mrow[r] = -1e30f; lrow[r] = 0.0f; }
    const float rscale = 0.08838834764831845f;   // 1/sqrt(128)

    for (int kb = 0; kb < SEQ; kb += KT) {
        __syncthreads();   // previous iteration done reading ldsK/ldsV
        // commit prefetched K/V to LDS (V transposed -> [d][key])
        #pragma unroll
        for (int it = 0; it < 8; ++it) {
            const int i  = tid + it * 128;
            const int r  = i >> 4;
            const int c8 = (i & 15) * 8;
            *(u32x4*)(ldsK + r * LK + c8) = rk[it];
            union { u32x4 u; bf16_t h8[8]; } u;
            u.u = rv[it];
            #pragma unroll
            for (int j = 0; j < 8; ++j)
                ldsV[(c8 + j) * LV + r] = u.h8[j];
        }
        __syncthreads();
        if (kb + KT < SEQ) preload(kb + KT);

        // ---- scores: 16 q rows x 64 keys per wave ----
        f32x8 sacc[4] = {};
        #pragma unroll
        for (int kd = 0; kd < 4; ++kd) {
            bf16x16 a = load_frag_a(ldsQ, LQ, wave * 16, kd * 32, lane);
            #pragma unroll
            for (int nt = 0; nt < 4; ++nt) {
                bf16x16 b = load_frag_b(ldsK, LK, nt * 16, kd * 32, lane);
                sacc[nt] = wmma_bf16(a, b, sacc[nt]);
            }
        }

        // ---- online softmax (per lane: 8 rows of its half) ----
        float scaleo[8];
        #pragma unroll
        for (int r = 0; r < 8; ++r) {
            float mx = -1e30f;
            #pragma unroll
            for (int nt = 0; nt < 4; ++nt)
                mx = fmaxf(mx, sacc[nt][r] * rscale);
            #pragma unroll
            for (int sh = 1; sh < 16; sh <<= 1)
                mx = fmaxf(mx, __shfl_xor(mx, sh, 32));
            const float nm = fmaxf(mrow[r], mx);
            scaleo[r] = __expf(mrow[r] - nm);
            float rs = 0.0f;
            #pragma unroll
            for (int nt = 0; nt < 4; ++nt) {
                const float p0 = __expf(sacc[nt][r] * rscale - nm);
                sacc[nt][r] = p0;
                rs += p0;
            }
            #pragma unroll
            for (int sh = 1; sh < 16; sh <<= 1)
                rs += __shfl_xor(rs, sh, 32);
            lrow[r] = lrow[r] * scaleo[r] + rs;
            mrow[r] = nm;
        }
        #pragma unroll
        for (int t = 0; t < 8; ++t)
            #pragma unroll
            for (int r = 0; r < 8; ++r)
                oacc[t][r] *= scaleo[r];

        // ---- P (C-layout) -> LDS (A-layout source) ----
        {
            bf16_t* pw = ldsP + wave * 16 * LP;
            const int half = lane >> 4;
            const int nr   = lane & 15;
            #pragma unroll
            for (int nt = 0; nt < 4; ++nt)
                #pragma unroll
                for (int r = 0; r < 8; ++r)
                    pw[(half * 8 + r) * LP + nt * 16 + nr] = (bf16_t)sacc[nt][r];
        }
        __syncthreads();

        // ---- O += P @ V ----
        const bf16_t* pw = ldsP + wave * 16 * LP;
        #pragma unroll
        for (int kd = 0; kd < 2; ++kd) {
            bf16x16 a = load_frag_a(pw, LP, 0, kd * 32, lane);
            #pragma unroll
            for (int nt = 0; nt < 8; ++nt) {
                bf16x16 b = load_frag_b(ldsV, LV, nt * 16, kd * 32, lane);
                oacc[nt] = wmma_bf16(a, b, oacc[nt]);
            }
        }
    }

    // ---- epilogue: normalize, store bf16 ----
    float rinv[8];
    #pragma unroll
    for (int r = 0; r < 8; ++r) rinv[r] = 1.0f / lrow[r];
    const int half = lane >> 4;
    const int nr   = lane & 15;
    #pragma unroll
    for (int nt = 0; nt < 8; ++nt) {
        #pragma unroll
        for (int r = 0; r < 8; ++r) {
            const int row = q0 + wave * 16 + half * 8 + r;
            ab[(size_t)row * DMODEL + headOff + nt * 16 + nr] =
                (bf16_t)(oacc[nt][r] * rinv[r]);
        }
    }
}

// ---------------------------------------------------------------------------
// Launch
// ---------------------------------------------------------------------------
extern "C" void kernel_launch(void* const* d_in, const int* in_sizes, int n_in,
                              void* d_out, int out_size, void* d_ws, size_t ws_size,
                              hipStream_t stream) {
    const float* x    = (const float*)d_in[0];
    const float* fcos = (const float*)d_in[1];
    const float* fsin = (const float*)d_in[2];
    const float* wq   = (const float*)d_in[3];
    const float* wk   = (const float*)d_in[4];
    const float* wv   = (const float*)d_in[5];
    const float* wo   = (const float*)d_in[6];
    float* out = (float*)d_out;

    const size_t DD = (size_t)DMODEL * DMODEL;   // 16.78M elems
    const size_t SD = (size_t)SEQ * DMODEL;      //  8.39M elems
    bf16_t* wqT = (bf16_t*)d_ws;
    bf16_t* wkT = wqT + DD;
    bf16_t* wvT = wkT + DD;
    bf16_t* woT = wvT + DD;
    bf16_t* xb  = woT + DD;
    bf16_t* qb  = xb + SD;
    bf16_t* kb  = qb + SD;
    bf16_t* vb  = kb + SD;
    bf16_t* ab  = vb + SD;

    // 1) one-time conversions
    cvt_f32_bf16<<<(SD / 8 + 255) / 256, 256, 0, stream>>>(x, xb);
    dim3 tgrid(DMODEL / 64, DMODEL / 64);
    transpose_cvt<<<tgrid, 256, 0, stream>>>(wq, wqT, DMODEL, DMODEL);
    transpose_cvt<<<tgrid, 256, 0, stream>>>(wk, wkT, DMODEL, DMODEL);
    transpose_cvt<<<tgrid, 256, 0, stream>>>(wv, wvT, DMODEL, DMODEL);
    transpose_cvt<<<tgrid, 256, 0, stream>>>(wo, woT, DMODEL, DMODEL);

    // 2) projections (RoPE fused into Q/K epilogues)
    dim3 ggrid(DMODEL / GBN, SEQ / GBM);         // (32, 16)
    gemm_bf16<2><<<ggrid, 256, 0, stream>>>(xb, wqT, qb, fcos, fsin, SEQ, DMODEL, DMODEL);
    gemm_bf16<2><<<ggrid, 256, 0, stream>>>(xb, wkT, kb, fcos, fsin, SEQ, DMODEL, DMODEL);
    gemm_bf16<1><<<ggrid, 256, 0, stream>>>(xb, wvT, vb, nullptr, nullptr, SEQ, DMODEL, DMODEL);

    // 3) attention
    attn_kernel<<<dim3(SEQ / QT, NUM_HEADS), 128, 0, stream>>>(qb, kb, vb, ab);

    // 4) output projection
    gemm_bf16<0><<<ggrid, 256, 0, stream>>>(ab, woT, out, nullptr, nullptr, SEQ, DMODEL, DMODEL);
}
